// MambaLayer_70755291234632
// MI455X (gfx1250) — compile-verified, matrix operand order checked by reading, not run on hardware
//
#include <hip/hip_runtime.h>

// Mamba layer: B=1, L=2048, DM=2048, DI=4096, DS=16, DTR=128, DC=4
#define DI_   4096
#define DM_   2048
#define LSEQ  2048
#define DS_   16
#define DTR_  128
#define DC_   4
#define E2_   (2*DI_)        // 8192
#define PDIM  (DTR_ + 2*DS_) // 160
#define NC    32             // scan chunks
#define CHUNK (LSEQ / NC)    // 64 steps per chunk

typedef __attribute__((ext_vector_type(16))) __bf16 v16bf;
typedef __attribute__((ext_vector_type(8)))  __bf16 v8bf;
typedef __attribute__((ext_vector_type(8)))  float  v8f;
typedef __attribute__((ext_vector_type(4)))  float  v4f;
typedef int v4i_vs __attribute__((__vector_size__(4 * sizeof(int))));

#define BM 128
#define BN 128
#define BK 32
#define APITCH 40   // bf16 units; rows stay 16B-aligned, staggers banks

#define ACT_NONE 0
#define ACT_SOFTPLUS_BIAS 1

// ---------------------------------------------------------------------------
// C[m*cs_m + n*cs_n] = epilogue( sum_k A[m*lda+k] * B[n*ldb+k] )
// f32 inputs converted to bf16 in LDS; v_wmma_f32_16x16x32_bf16 compute.
// Block tile 128x128x32; 8 wave32, each wave owns 32x64 (8 WMMA / K-step).
// Grid: (N/128, ceil(M/128)), block 256. N mult of 128, K mult of 32.
// ---------------------------------------------------------------------------
__global__ __launch_bounds__(256) void gemm_tn_bf16(
    const float* __restrict__ A, const float* __restrict__ B, float* __restrict__ C,
    int M, int N, int K, int lda, int ldb, long cs_m, long cs_n,
    const float* __restrict__ bias, int act)
{
  __shared__ __bf16 As[BM * APITCH];
  __shared__ __bf16 Bs[BN * APITCH];

  const int tid    = threadIdx.x;
  const int tile_m = blockIdx.y * BM;
  const int tile_n = blockIdx.x * BN;
  const int wave   = tid >> 5;
  const int lane   = tid & 31;
  const int hi     = lane >> 4;   // half-wave select
  const int lr     = lane & 15;
  const int wm     = wave & 3;    // wave's 32-row strip
  const int wn     = wave >> 2;   // wave's 64-col group

  v8f acc[2][4];
  #pragma unroll
  for (int i = 0; i < 2; ++i)
    #pragma unroll
    for (int j = 0; j < 4; ++j) acc[i][j] = (v8f){};

  const int srow = tid >> 1;        // 0..127
  const int scol = (tid & 1) * 16;  // 0 or 16

  for (int k0 = 0; k0 < K; k0 += BK) {
    // --- stage panels (f32 -> bf16), 16 values per thread per panel ---
    {
      v8bf ta0, ta1;
      const int gr = tile_m + srow;
      if (gr < M) {
        const float* src = A + (long)gr * lda + k0 + scol;
        if (k0 + BK < K) __builtin_prefetch(src + BK, 0, 3);
        #pragma unroll
        for (int i = 0; i < 8; ++i) { ta0[i] = (__bf16)src[i]; ta1[i] = (__bf16)src[8 + i]; }
      } else {
        #pragma unroll
        for (int i = 0; i < 8; ++i) { ta0[i] = (__bf16)0.0f; ta1[i] = (__bf16)0.0f; }
      }
      *(v8bf*)&As[srow * APITCH + scol]     = ta0;
      *(v8bf*)&As[srow * APITCH + scol + 8] = ta1;

      const float* srcb = B + (long)(tile_n + srow) * ldb + k0 + scol;
      if (k0 + BK < K) __builtin_prefetch(srcb + BK, 0, 3);
      v8bf tb0, tb1;
      #pragma unroll
      for (int i = 0; i < 8; ++i) { tb0[i] = (__bf16)srcb[i]; tb1[i] = (__bf16)srcb[8 + i]; }
      *(v8bf*)&Bs[srow * APITCH + scol]     = tb0;
      *(v8bf*)&Bs[srow * APITCH + scol + 8] = tb1;
    }
    __syncthreads();

    // --- fragments per ISA 16-bit layout:
    // lanes 0-15: K = hi*8+{0..7} in v0..3, 16+hi*8+{0..7} in v4..7
    v16bf afrag[2], bfrag[4];
    #pragma unroll
    for (int i = 0; i < 2; ++i) {
      const __bf16* ap = &As[(wm * 32 + i * 16 + lr) * APITCH];
      v8bf lo = *(const v8bf*)&ap[hi * 8];
      v8bf hi8 = *(const v8bf*)&ap[16 + hi * 8];
      #pragma unroll
      for (int e = 0; e < 8; ++e) { afrag[i][e] = lo[e]; afrag[i][8 + e] = hi8[e]; }
    }
    #pragma unroll
    for (int j = 0; j < 4; ++j) {
      const __bf16* bp = &Bs[(wn * 64 + j * 16 + lr) * APITCH];
      v8bf lo = *(const v8bf*)&bp[hi * 8];
      v8bf hi8 = *(const v8bf*)&bp[16 + hi * 8];
      #pragma unroll
      for (int e = 0; e < 8; ++e) { bfrag[j][e] = lo[e]; bfrag[j][8 + e] = hi8[e]; }
    }

    #pragma unroll
    for (int i = 0; i < 2; ++i)
      #pragma unroll
      for (int j = 0; j < 4; ++j)
        acc[i][j] = __builtin_amdgcn_wmma_f32_16x16x32_bf16(
            false, afrag[i], false, bfrag[j], (short)0, acc[i][j], false, false);
    __syncthreads();
  }

  // epilogue: lane lr = N col, VGPR r = row (r + 8*hi); strided store
  #pragma unroll
  for (int i = 0; i < 2; ++i) {
    #pragma unroll
    for (int r = 0; r < 8; ++r) {
      const int gm = tile_m + wm * 32 + i * 16 + hi * 8 + r;
      if (gm < M) {
        #pragma unroll
        for (int j = 0; j < 4; ++j) {
          const int gn = tile_n + wn * 64 + j * 16 + lr;
          float v = acc[i][j][r];
          if (act == ACT_SOFTPLUS_BIAS) {
            v += bias[gm];
            v = (v > 20.f) ? v : log1pf(__expf(v));
          }
          C[(long)gm * cs_m + (long)gn * cs_n] = v;
        }
      }
    }
  }
}

// ---------------------------------------------------------------------------
// Causal depthwise conv1d (kernel 4) + SiLU.
// Input x = xz[:, :DI] in (L, 2DI) layout; output xc in (L, DI) layout.
// ---------------------------------------------------------------------------
__global__ __launch_bounds__(256) void conv_silu_kernel(
    const float* __restrict__ xz, const float* __restrict__ w, float* __restrict__ xc)
{
  long idx = (long)blockIdx.x * blockDim.x + threadIdx.x;
  if (idx >= (long)LSEQ * DI_) return;
  const int d = (int)(idx % DI_);
  const int l = (int)(idx / DI_);
  float acc = 0.f;
  #pragma unroll
  for (int k = 0; k < DC_; ++k) {
    const int ls = l + k - (DC_ - 1);
    const float v = (ls >= 0) ? xz[(long)ls * E2_ + d] : 0.f;
    acc += v * w[d * DC_ + k];
  }
  xc[idx] = acc / (1.f + __expf(-acc));   // silu
}

// --- gfx1250 async broadcast of B_t/C_t (32 floats) into LDS ---------------
__device__ __forceinline__ void load_bc_async(const float* __restrict__ x_dbl,
                                              int t, float* sBC) {
#if __has_builtin(__builtin_amdgcn_global_load_async_to_lds_b128)
  if (threadIdx.x < 8) {
    const float* gsrc = &x_dbl[(long)t * PDIM + DTR_ + threadIdx.x * 4];
    __builtin_amdgcn_global_load_async_to_lds_b128(
        (__attribute__((address_space(1))) v4i_vs*)gsrc,
        (__attribute__((address_space(3))) v4i_vs*)&sBC[threadIdx.x * 4],
        0, 0);
  }
#if __has_builtin(__builtin_amdgcn_s_wait_asynccnt)
  __builtin_amdgcn_s_wait_asynccnt(0);
#else
  asm volatile("s_wait_asynccnt 0x0" ::: "memory");
#endif
#else
  if (threadIdx.x < 32) sBC[threadIdx.x] = x_dbl[(long)t * PDIM + DTR_ + threadIdx.x];
#endif
}

// ---------------------------------------------------------------------------
// Chunked parallel selective scan (3 passes). Recurrence per (d,n):
//   h_t = dA_t * h_{t-1} + (dt_t * u_t) * B_t[n],  dA_t = exp(dt_t * A[d][n])
// Pass 1: per chunk, compute decay product P_c and local final state H_c (h0=0).
// ---------------------------------------------------------------------------
__global__ __launch_bounds__(256) void scan_pass1(
    const float* __restrict__ delta, const float* __restrict__ xc,
    const float* __restrict__ x_dbl, const float* __restrict__ A_log,
    float* __restrict__ P, float* __restrict__ Hc)
{
  __shared__ __align__(16) float sBC[32];
  const int d = blockIdx.x * 256 + threadIdx.x;
  const int c = blockIdx.y;
  const int t0 = c * CHUNK;

  float a[DS_], h[DS_], p[DS_];
  #pragma unroll
  for (int n = 0; n < DS_; ++n) { a[n] = -__expf(A_log[d * DS_ + n]); h[n] = 0.f; p[n] = 1.f; }

  for (int tt = 0; tt < CHUNK; ++tt) {
    const int t = t0 + tt;
    __syncthreads();
    load_bc_async(x_dbl, t, sBC);
    __syncthreads();

    const float dt = delta[(long)t * DI_ + d];
    const float u  = xc[(long)t * DI_ + d];
    if (tt + 1 < CHUNK) {
      __builtin_prefetch(&delta[(long)(t + 1) * DI_ + d], 0, 3);
      __builtin_prefetch(&xc[(long)(t + 1) * DI_ + d], 0, 3);
    }
    const float du = dt * u;
    #pragma unroll
    for (int n = 0; n < DS_; ++n) {
      const float dA = __expf(dt * a[n]);
      h[n] = h[n] * dA + du * sBC[n];
      p[n] *= dA;
    }
  }

  float* pp = &P[((long)c * DI_ + d) * DS_];
  float* hp = &Hc[((long)c * DI_ + d) * DS_];
  #pragma unroll
  for (int n = 0; n < DS_; n += 4) {
    *(v4f*)&pp[n] = (v4f){p[n], p[n+1], p[n+2], p[n+3]};
    *(v4f*)&hp[n] = (v4f){h[n], h[n+1], h[n+2], h[n+3]};
  }
}

// Pass 2: sequential combine across NC chunks, parallel over all 65536 (d,n)
// pairs. In place: Hc[c] becomes the TRUE initial state for chunk c.
__global__ __launch_bounds__(256) void scan_combine(
    const float* __restrict__ P, float* __restrict__ Hc)
{
  const long pair = (long)blockIdx.x * 256 + threadIdx.x;  // (d,n) pair
  const long stride = (long)DI_ * DS_;
  float running = 0.f;
  for (int c = 0; c < NC; ++c) {
    const float tmp = Hc[c * stride + pair];
    const float pp  = P[c * stride + pair];
    Hc[c * stride + pair] = running;
    running = running * pp + tmp;
  }
}

// Pass 3: local scan seeded with true h0; y = (sum_n h*C + u*D) * silu(z).
__global__ __launch_bounds__(256) void scan_pass3(
    const float* __restrict__ delta, const float* __restrict__ xc,
    const float* __restrict__ x_dbl, const float* __restrict__ xz,
    const float* __restrict__ A_log, const float* __restrict__ Dp,
    const float* __restrict__ H0, float* __restrict__ Y)
{
  __shared__ __align__(16) float sBC[32];
  const int d = blockIdx.x * 256 + threadIdx.x;
  const int c = blockIdx.y;
  const int t0 = c * CHUNK;

  float a[DS_], h[DS_];
  const float* h0p = &H0[((long)c * DI_ + d) * DS_];
  #pragma unroll
  for (int n = 0; n < DS_; ++n) { a[n] = -__expf(A_log[d * DS_ + n]); h[n] = h0p[n]; }
  const float Dd = Dp[d];

  for (int tt = 0; tt < CHUNK; ++tt) {
    const int t = t0 + tt;
    __syncthreads();
    load_bc_async(x_dbl, t, sBC);
    __syncthreads();

    const float dt = delta[(long)t * DI_ + d];
    const float u  = xc[(long)t * DI_ + d];
    if (tt + 1 < CHUNK) {
      __builtin_prefetch(&delta[(long)(t + 1) * DI_ + d], 0, 3);
      __builtin_prefetch(&xc[(long)(t + 1) * DI_ + d], 0, 3);
      __builtin_prefetch(&xz[(long)(t + 1) * E2_ + DI_ + d], 0, 3);
    }
    const float du = dt * u;
    float y = 0.f;
    #pragma unroll
    for (int n = 0; n < DS_; ++n) {
      const float dA = __expf(dt * a[n]);
      h[n] = h[n] * dA + du * sBC[n];
      y += h[n] * sBC[16 + n];
    }
    const float z = xz[(long)t * E2_ + DI_ + d];
    y = (y + u * Dd) * (z / (1.f + __expf(-z)));
    Y[(long)t * DI_ + d] = y;
  }
}

extern "C" void kernel_launch(void* const* d_in, const int* in_sizes, int n_in,
                              void* d_out, int out_size, void* d_ws, size_t ws_size,
                              hipStream_t stream) {
  const float* hidden     = (const float*)d_in[0]; // (1, L, DM)
  const float* in_proj_w  = (const float*)d_in[1]; // (2DI, DM)
  const float* conv_w     = (const float*)d_in[2]; // (DI, 1, DC)
  const float* x_proj_w   = (const float*)d_in[3]; // (160, DI)
  const float* dt_proj_w  = (const float*)d_in[4]; // (DI, DTR)
  const float* out_proj_w = (const float*)d_in[5]; // (DM, DI)
  const float* dt_bias    = (const float*)d_in[6]; // (DI,)
  const float* A_log      = (const float*)d_in[7]; // (DI, DS)
  const float* Dvec       = (const float*)d_in[8]; // (DI,)
  float* out = (float*)d_out;                      // (1, L, DM)

  char* ws = (char*)d_ws;
  float* xz    = (float*)ws; ws += sizeof(float) * (size_t)LSEQ * E2_;       // (L, 2DI)
  float* xc    = (float*)ws; ws += sizeof(float) * (size_t)LSEQ * DI_;       // (L, DI)
  float* x_dbl = (float*)ws; ws += sizeof(float) * (size_t)LSEQ * PDIM;      // (L, 160)
  float* delta = (float*)ws; ws += sizeof(float) * (size_t)LSEQ * DI_;       // (L, DI)
  float* Yb    = (float*)ws; ws += sizeof(float) * (size_t)LSEQ * DI_;       // (L, DI)
  float* Pbuf  = (float*)ws; ws += sizeof(float) * (size_t)NC * DI_ * DS_;   // (NC, DI, DS)
  float* Hcbuf = (float*)ws; ws += sizeof(float) * (size_t)NC * DI_ * DS_;   // (NC, DI, DS)

  const dim3 blk(256);
  const long ne = (long)LSEQ * DI_;

  // 1) in_proj: xz[l][e] = sum_d W[e][d] * hidden[l][d]
  gemm_tn_bf16<<<dim3(LSEQ / BN, E2_ / BM), blk, 0, stream>>>(
      in_proj_w, hidden, xz, E2_, LSEQ, DM_, DM_, DM_, 1L, (long)E2_, nullptr, ACT_NONE);

  // 2) depthwise causal conv + silu -> xc (L, DI)
  conv_silu_kernel<<<dim3((unsigned)((ne + 255) / 256)), blk, 0, stream>>>(xz, conv_w, xc);

  // 3) x_proj: x_dbl[l][p] = sum_d W[p][d] * xc[l][d]   (M=160, guarded)
  gemm_tn_bf16<<<dim3(LSEQ / BN, (PDIM + BM - 1) / BM), blk, 0, stream>>>(
      x_proj_w, xc, x_dbl, PDIM, LSEQ, DI_, DI_, DI_, 1L, (long)PDIM, nullptr, ACT_NONE);

  // 4) dt_proj + fused bias/softplus: delta[l][d] = softplus(W dt + bias)
  gemm_tn_bf16<<<dim3(LSEQ / BN, DI_ / BM), blk, 0, stream>>>(
      dt_proj_w, x_dbl, delta, DI_, LSEQ, DTR_, DTR_, PDIM, 1L, (long)DI_,
      dt_bias, ACT_SOFTPLUS_BIAS);

  // 5) chunked parallel selective scan (serial depth 2048 -> 64)
  scan_pass1<<<dim3(DI_ / 256, NC), blk, 0, stream>>>(
      delta, xc, x_dbl, A_log, Pbuf, Hcbuf);
  scan_combine<<<dim3((DI_ * DS_) / 256), blk, 0, stream>>>(Pbuf, Hcbuf);
  scan_pass3<<<dim3(DI_ / 256, NC), blk, 0, stream>>>(
      delta, xc, x_dbl, xz, A_log, Dvec, Hcbuf, Yb);

  // 6) out_proj: out[l][m] = sum_d W[m][d] * Y[l][d]
  gemm_tn_bf16<<<dim3(LSEQ / BN, DM_ / BM), blk, 0, stream>>>(
      out_proj_w, Yb, out, DM_, LSEQ, DI_, DI_, DI_, 1L, (long)DM_, nullptr, ACT_NONE);
}
